// Decoder_19799799235208
// MI455X (gfx1250) — matrix operand order repaired and checked by reading, not compiled, not run on hardware
//
#include <hip/hip_runtime.h>
#include <hip/hip_bf16.h>

// ---------------------------------------------------------------------------
// MI455X (gfx1250) decoder layer: all matmuls via v_wmma_f32_16x16x32_bf16,
// tiles staged with GLOBAL_LOAD_ASYNC_TO_LDS_B128 (ASYNCcnt) + LDS double
// buffering. B=4, S=T=1024, D=512, H=8. ~326 GFLOP vs ~0.5GB -> compute bound.
// Epilogue flags are template params so the accumulators never spill.
// ---------------------------------------------------------------------------

typedef __attribute__((ext_vector_type(16))) __bf16 v16bf;
typedef __attribute__((ext_vector_type(8)))  float  v8f;
typedef __attribute__((ext_vector_type(4)))  int    v4i_;

#define AS1 __attribute__((address_space(1)))
#define AS3 __attribute__((address_space(3)))

#define FLAG_BIAS 1
#define FLAG_RELU 2
#define FLAG_ST   4   // store C transposed: C[n*ldc + m]
#define FLAG_OBF  8   // output bf16 (else f32)

#define GPITCH 72     // LDS row pitch in bf16 elems (64 K + 8 pad -> bank spread)
#define BKG    64
#define LTILE  (128 * GPITCH)             // elems per staged matrix tile
#define GEMM_LDS_BYTES (4 * LTILE * 2)    // 2 buffers x (A + B) x bf16

#if __has_builtin(__builtin_amdgcn_global_load_async_to_lds_b128)
#define HAVE_ASYNC_LDS 1
#else
#define HAVE_ASYNC_LDS 0
#endif

// 16B global -> LDS copy; async (direct-to-LDS, ASYNCcnt) when available.
__device__ __forceinline__ void copy16_g2l(const __bf16* g, __bf16* l) {
#if HAVE_ASYNC_LDS
  __builtin_amdgcn_global_load_async_to_lds_b128(
      (AS1 v4i_*)(AS1 void*)g,
      (AS3 v4i_*)(AS3 void*)l, 0, 0);
#else
  *(uint4*)l = *(const uint4*)g;
#endif
}

__device__ __forceinline__ void wait_async_all() {
#if HAVE_ASYNC_LDS
#if __has_builtin(__builtin_amdgcn_s_wait_asynccnt)
  __builtin_amdgcn_s_wait_asynccnt(0);
#else
  asm volatile("s_wait_asynccnt 0" ::: "memory");
#endif
#endif
}

struct FragU { union { v16bf v; uint4 u[2]; }; };

// C[M,N] = alpha * A[M,K] * B[N,K]^T (+ bias[n]) ; per-z offsets:
//   off = (z/zdiv)*Out + (z%zdiv)*In  for A, B, C;  bias += (z%zdiv)*biasIn
template <int FLAGS>
__global__ __launch_bounds__(256) void gemm_nt_bf16(
    const __bf16* __restrict__ A, const __bf16* __restrict__ Bm,
    void* __restrict__ C, const float* __restrict__ bias,
    int K, int lda, int ldb, int ldc,
    long long aOut, long long aIn, long long bOut, long long bIn,
    long long cOut, long long cIn, int zdiv, int biasIn,
    float alpha)
{
  extern __shared__ __bf16 smem[];   // [2 bufs][A tile | B tile]

  const int z  = blockIdx.z;
  const int zo = z / zdiv;
  const int zi = z - zo * zdiv;
  A  += zo * aOut + zi * aIn;
  Bm += zo * bOut + zi * bIn;
  const long long cBase = zo * cOut + zi * cIn;
  const float* bptr = (FLAGS & FLAG_BIAS) ? (bias + (long long)zi * biasIn) : nullptr;

  const int tid  = threadIdx.x;
  const int wid  = tid >> 5;
  const int lane = tid & 31;
  const int hf   = lane >> 4;
  const int l15  = lane & 15;
  const int waveM = wid >> 1;      // 0..3
  const int waveN = wid & 1;       // 0..1
  const int mBase = blockIdx.y * 128;
  const int nBase = blockIdx.x * 128;

  // stage 128x64 bf16 tiles of A and B into LDS buffer p (4x 16B chunks/thread)
  auto stage = [&](int kb, int p) {
    __bf16* dA = smem + p * 2 * LTILE;
    __bf16* dB = dA + LTILE;
#pragma unroll
    for (int i = 0; i < 4; ++i) {
      int c   = tid + (i << 8);
      int row = c >> 3;
      int col = (c & 7) << 3;
      copy16_g2l(&A[(long long)(mBase + row) * lda + kb * BKG + col],
                 &dA[row * GPITCH + col]);
      copy16_g2l(&Bm[(long long)(nBase + row) * ldb + kb * BKG + col],
                 &dB[row * GPITCH + col]);
#if !HAVE_ASYNC_LDS
      __builtin_prefetch(&A[(long long)(mBase + row) * lda + (kb + 1) * BKG + col], 0, 1);
      __builtin_prefetch(&Bm[(long long)(nBase + row) * ldb + (kb + 1) * BKG + col], 0, 1);
#endif
    }
  };

  v8f acc[2][4];
#pragma unroll
  for (int i = 0; i < 2; ++i)
#pragma unroll
    for (int j = 0; j < 4; ++j)
#pragma unroll
      for (int r = 0; r < 8; ++r) acc[i][j][r] = 0.0f;

  const int nk = K / BKG;
  stage(0, 0);
  wait_async_all();
  __syncthreads();

  for (int kb = 0; kb < nk; ++kb) {
    const int p = kb & 1;
    if (kb + 1 < nk) stage(kb + 1, p ^ 1);   // async fill of next buffer

    const __bf16* sA = smem + p * 2 * LTILE;
    const __bf16* sB = sA + LTILE;
#pragma unroll
    for (int kk = 0; kk < BKG; kk += 32) {
      FragU af[2], bfb[4];
      // A frag (16x32 bf16): lane row = l15; K chunks {hf*8..} and {16+hf*8..}
#pragma unroll
      for (int mi = 0; mi < 2; ++mi) {
        int rb = (waveM * 32 + mi * 16 + l15) * GPITCH + kk;
        af[mi].u[0] = *(const uint4*)&sA[rb + hf * 8];
        af[mi].u[1] = *(const uint4*)&sA[rb + 16 + hf * 8];
      }
      // B frag (32x16 bf16): lane col = l15; K chunks {hf*8..} and {16+hf*8..}
#pragma unroll
      for (int ni = 0; ni < 4; ++ni) {
        int rb = (waveN * 64 + ni * 16 + l15) * GPITCH + kk;
        bfb[ni].u[0] = *(const uint4*)&sB[rb + hf * 8];
        bfb[ni].u[1] = *(const uint4*)&sB[rb + 16 + hf * 8];
      }
#pragma unroll
      for (int mi = 0; mi < 2; ++mi)
#pragma unroll
        for (int ni = 0; ni < 4; ++ni)
          acc[mi][ni] = __builtin_amdgcn_wmma_f32_16x16x32_bf16(
              false, af[mi].v, false, bfb[ni].v, (short)0, acc[mi][ni], false, false);
    }

    if (kb + 1 < nk) {
      wait_async_all();     // my wave's async fills landed
      __syncthreads();      // everyone's landed; also fences compute of buf p
    }
  }

  // Epilogue (fully unrolled, compile-time flags -> no acc spill):
  // C VGPR r holds M = r + 8*hf, N = l15 within each 16x16 tile
#pragma unroll
  for (int mi = 0; mi < 2; ++mi)
#pragma unroll
    for (int ni = 0; ni < 4; ++ni) {
      int mg = mBase + waveM * 32 + mi * 16 + hf * 8;
      int ng = nBase + waveN * 64 + ni * 16 + l15;
      float bvv = (FLAGS & FLAG_BIAS) ? bptr[ng] : 0.0f;
#pragma unroll
      for (int r = 0; r < 8; ++r) {
        float v = acc[mi][ni][r] * alpha + bvv;
        if (FLAGS & FLAG_RELU) v = fmaxf(v, 0.0f);
        long long m = mg + r;
        long long idx = cBase + ((FLAGS & FLAG_ST) ? ((long long)ng * ldc + m)
                                                   : (m * ldc + (long long)ng));
        if (FLAGS & FLAG_OBF) ((__bf16*)C)[idx] = (__bf16)v;
        else                  ((float*)C)[idx]  = v;
      }
    }
}

// f32 -> bf16 elementwise
__global__ void cvt_bf16(const float* __restrict__ in, __bf16* __restrict__ out, long long n) {
  long long i = (long long)blockIdx.x * blockDim.x + threadIdx.x;
  if (i < n) out[i] = (__bf16)in[i];
}

// Batched f32 -> bf16 transpose: out[b][c][r] = in[b][r][c]; rows,cols % 32 == 0
__global__ void cvtT_bf16(const float* __restrict__ in, __bf16* __restrict__ out,
                          int rows, int cols) {
  __shared__ float t[32][33];
  long long base = (long long)blockIdx.z * rows * cols;
  int c0 = blockIdx.x * 32, r0 = blockIdx.y * 32;
  int tx = threadIdx.x, ty = threadIdx.y;  // block (32,8)
  for (int j = 0; j < 32; j += 8)
    t[ty + j][tx] = in[base + (long long)(r0 + ty + j) * cols + c0 + tx];
  __syncthreads();
  for (int j = 0; j < 32; j += 8)
    out[base + (long long)(c0 + ty + j) * rows + r0 + tx] = (__bf16)t[tx][ty + j];
}

// Row softmax (optionally causal), f32 scores -> bf16 probs
__global__ void softmax_rows(const float* __restrict__ S, __bf16* __restrict__ P,
                             int T, int Slen, int causal) {
  __shared__ float red[256];
  long long row = blockIdx.x;
  int s = (int)(row % Slen);
  const float* p = S + row * T;
  __bf16* q = P + row * T;
  int tid = threadIdx.x;
  int lim = causal ? (s + 1) : T;

  float mx = -3.0e38f;
  for (int i = tid; i < lim; i += 256) mx = fmaxf(mx, p[i]);
  red[tid] = mx; __syncthreads();
  for (int o = 128; o > 0; o >>= 1) {
    if (tid < o) red[tid] = fmaxf(red[tid], red[tid + o]);
    __syncthreads();
  }
  mx = red[0]; __syncthreads();

  float sum = 0.0f;
  for (int i = tid; i < lim; i += 256) sum += __expf(p[i] - mx);
  red[tid] = sum; __syncthreads();
  for (int o = 128; o > 0; o >>= 1) {
    if (tid < o) red[tid] += red[tid + o];
    __syncthreads();
  }
  float inv = 1.0f / red[0];

  for (int i = tid; i < T; i += 256) {
    float v = (i < lim) ? __expf(p[i] - mx) * inv : 0.0f;
    q[i] = (__bf16)v;
  }
}

// out = LayerNorm(a + b) * g + beta ; also optional bf16 copy for next GEMM
__global__ void add_ln(const float* __restrict__ a, const float* __restrict__ b,
                       const float* __restrict__ g, const float* __restrict__ be,
                       float* __restrict__ out, __bf16* __restrict__ outbf, int Dd) {
  __shared__ float r1[256];
  __shared__ float r2[256];
  long long row = blockIdx.x;
  int tid = threadIdx.x;
  const float* pa = a + row * Dd;
  const float* pb = b + row * Dd;
  float s = 0.0f, s2 = 0.0f;
  for (int i = tid; i < Dd; i += 256) {
    float v = pa[i] + pb[i];
    s += v; s2 += v * v;
  }
  r1[tid] = s; r2[tid] = s2; __syncthreads();
  for (int o = 128; o > 0; o >>= 1) {
    if (tid < o) { r1[tid] += r1[tid + o]; r2[tid] += r2[tid + o]; }
    __syncthreads();
  }
  float mean = r1[0] / Dd;
  float var  = r2[0] / Dd - mean * mean;
  float rstd = rsqrtf(var + 1e-5f);
  for (int i = tid; i < Dd; i += 256) {
    float v = pa[i] + pb[i];
    float y = (v - mean) * rstd * g[i] + be[i];
    out[row * Dd + i] = y;
    if (outbf) outbf[row * Dd + i] = (__bf16)y;
  }
}

extern "C" void kernel_launch(void* const* d_in, const int* in_sizes, int n_in,
                              void* d_out, int out_size, void* d_ws, size_t ws_size,
                              hipStream_t stream) {
  (void)in_sizes; (void)n_in; (void)out_size; (void)ws_size;
  const int B_ = 4, S_ = 1024, T_ = 1024, D_ = 512, H_ = 8;
  const long long BSD  = (long long)B_ * S_ * D_;
  const long long BHSD = (long long)B_ * H_ * S_ * D_;
  const long long BHST = (long long)B_ * H_ * S_ * T_;
  const long long HDD  = (long long)H_ * D_ * D_;
  const long long D4D  = (long long)D_ * 4 * D_;
  const long long BS4D = (long long)B_ * S_ * 4 * D_;

  // --- workspace arena ---
  char* wp = (char*)d_ws;
  auto alloc = [&](long long bytes) -> void* {
    void* p = (void*)wp;
    wp += (bytes + 255) & ~255ll;
    return p;
  };
  __bf16* xbf   = (__bf16*)alloc(BSD * 2);
  __bf16* encbf = (__bf16*)alloc(BSD * 2);
  __bf16* wT[10];                       // sa: q,k,v,o,f ; ca: q,k,v,o,f
  for (int i = 0; i < 10; ++i) wT[i] = (__bf16*)alloc(HDD * 2);
  __bf16* fc1T  = (__bf16*)alloc(D4D * 2);
  __bf16* fc2T  = (__bf16*)alloc(D4D * 2);
  __bf16* Qb    = (__bf16*)alloc(BHSD * 2);
  __bf16* Kb    = (__bf16*)alloc(BHSD * 2);
  __bf16* Vt    = (__bf16*)alloc(BHSD * 2);   // V^T per (b,h): [D,T]
  __bf16* Ob    = (__bf16*)alloc(BHSD * 2);
  __bf16* Ocat  = (__bf16*)alloc(BHSD * 2);   // [B,S,H*D]
  float*  scF   = (float*)alloc(BHST * 4);
  __bf16* prB   = (__bf16*)alloc(BHST * 2);
  float*  attnF = (float*)alloc(BSD * 4);
  float*  x1F   = (float*)alloc(BSD * 4);
  __bf16* x1bf  = (__bf16*)alloc(BSD * 2);
  float*  x2F   = (float*)alloc(BSD * 4);
  __bf16* x2bf  = (__bf16*)alloc(BSD * 2);
  __bf16* ff1   = (__bf16*)alloc(BS4D * 2);
  float*  ff2   = (float*)alloc(BSD * 4);

  // --- inputs ---
  const float* x_f   = (const float*)d_in[0];
  const float* enc_f = (const float*)d_in[1];
  const float* saW[10], * caW[10];
  for (int i = 0; i < 10; ++i) { saW[i] = (const float*)d_in[2 + i];
                                 caW[i] = (const float*)d_in[12 + i]; }
  // layout within saW/caW: 0 wq,1 bq,2 wk,3 bk,4 wv,5 bv,6 wo,7 bo,8 wf,9 bf
  const float* ln1g = (const float*)d_in[22]; const float* ln1b = (const float*)d_in[23];
  const float* ln2g = (const float*)d_in[24]; const float* ln2b = (const float*)d_in[25];
  const float* ln3g = (const float*)d_in[26]; const float* ln3b = (const float*)d_in[27];
  const float* fc1w = (const float*)d_in[28]; const float* fc1b = (const float*)d_in[29];
  const float* fc2w = (const float*)d_in[30]; const float* fc2b = (const float*)d_in[31];

  auto gemm = [&](const __bf16* A, const __bf16* Bm, void* C, const float* bias,
                  int M, int N, int K, int lda, int ldb, int ldc,
                  long long aOut, long long aIn, long long bOut, long long bIn,
                  long long cOut, long long cIn, int zdiv, int batch, int biasIn,
                  float alpha, int flags) {
    dim3 g(N / 128, M / 128, batch);
#define LAUNCH_GEMM(F)                                                         \
    gemm_nt_bf16<F><<<g, 256, GEMM_LDS_BYTES, stream>>>(                       \
        A, Bm, C, bias, K, lda, ldb, ldc, aOut, aIn, bOut, bIn, cOut, cIn,     \
        zdiv, biasIn, alpha)
    switch (flags) {
      case 0:                                   LAUNCH_GEMM(0); break;
      case FLAG_BIAS:                           LAUNCH_GEMM(FLAG_BIAS); break;
      case FLAG_OBF:                            LAUNCH_GEMM(FLAG_OBF); break;
      case FLAG_BIAS | FLAG_OBF:                LAUNCH_GEMM(FLAG_BIAS | FLAG_OBF); break;
      case FLAG_BIAS | FLAG_OBF | FLAG_ST:      LAUNCH_GEMM(FLAG_BIAS | FLAG_OBF | FLAG_ST); break;
      case FLAG_BIAS | FLAG_RELU | FLAG_OBF:    LAUNCH_GEMM(FLAG_BIAS | FLAG_RELU | FLAG_OBF); break;
      default: break;
    }
#undef LAUNCH_GEMM
  };

  // --- convert activations; transpose+convert weights ---
  cvt_bf16<<<(int)(BSD / 256), 256, 0, stream>>>(x_f, xbf, BSD);
  cvt_bf16<<<(int)(BSD / 256), 256, 0, stream>>>(enc_f, encbf, BSD);
  const int wIdx[5] = {0, 2, 4, 6, 8};
  for (int i = 0; i < 4; ++i) {  // wq,wk,wv,wo  [H,D,D] -> per-head transpose
    cvtT_bf16<<<dim3(D_ / 32, D_ / 32, H_), dim3(32, 8), 0, stream>>>(saW[wIdx[i]], wT[i], D_, D_);
    cvtT_bf16<<<dim3(D_ / 32, D_ / 32, H_), dim3(32, 8), 0, stream>>>(caW[wIdx[i]], wT[5 + i], D_, D_);
  }
  cvtT_bf16<<<dim3(D_ / 32, (H_ * D_) / 32, 1), dim3(32, 8), 0, stream>>>(saW[8], wT[4], H_ * D_, D_);
  cvtT_bf16<<<dim3(D_ / 32, (H_ * D_) / 32, 1), dim3(32, 8), 0, stream>>>(caW[8], wT[9], H_ * D_, D_);
  cvtT_bf16<<<dim3((4 * D_) / 32, D_ / 32, 1), dim3(32, 8), 0, stream>>>(fc1w, fc1T, D_, 4 * D_);
  cvtT_bf16<<<dim3(D_ / 32, (4 * D_) / 32, 1), dim3(32, 8), 0, stream>>>(fc2w, fc2T, 4 * D_, D_);

  const float iscale = 0.044194173824159216f;  // 1/sqrt(512)

  auto mha = [&](const __bf16* xq, const __bf16* xkv, int Tl,
                 __bf16* const* w, const float* const* wb, int causal,
                 const float* resid, const float* lng, const float* lnb,
                 float* outF, __bf16* outBf) {
    // Q/K projections: [B*H] x ([S,D] @ [D,D]^T), per-head weight+bias
    gemm(xq, w[0], Qb, wb[0], S_, D_, D_, D_, D_, D_,
         (long long)S_ * D_, 0, 0, (long long)D_ * D_,
         (long long)H_ * S_ * D_, (long long)S_ * D_, H_, B_ * H_, D_, 1.0f,
         FLAG_BIAS | FLAG_OBF);
    gemm(xkv, w[1], Kb, wb[1], Tl, D_, D_, D_, D_, D_,
         (long long)Tl * D_, 0, 0, (long long)D_ * D_,
         (long long)H_ * Tl * D_, (long long)Tl * D_, H_, B_ * H_, D_, 1.0f,
         FLAG_BIAS | FLAG_OBF);
    // V projection stored TRANSPOSED -> Vt[b,h][D,Tl] so A*V is NT-contiguous
    gemm(xkv, w[2], Vt, wb[2], Tl, D_, D_, D_, D_, Tl,
         (long long)Tl * D_, 0, 0, (long long)D_ * D_,
         (long long)H_ * Tl * D_, (long long)Tl * D_, H_, B_ * H_, D_, 1.0f,
         FLAG_BIAS | FLAG_OBF | FLAG_ST);
    // scores = Q K^T / sqrt(D)   (f32)
    gemm(Qb, Kb, scF, nullptr, S_, Tl, D_, D_, D_, Tl,
         (long long)S_ * D_, 0, (long long)Tl * D_, 0,
         (long long)S_ * Tl, 0, 1, B_ * H_, 0, iscale, 0);
    softmax_rows<<<B_ * H_ * S_, 256, 0, stream>>>(scF, prB, Tl, S_, causal);
    // O = P V   (A=[S,Tl] bf16, Bt=V^T [D,Tl])
    gemm(prB, Vt, Ob, nullptr, S_, D_, Tl, Tl, Tl, D_,
         (long long)S_ * Tl, 0, (long long)Tl * D_, 0,
         (long long)S_ * D_, 0, 1, B_ * H_, 0, 1.0f, FLAG_OBF);
    // per-head out-linear, scattered into concat layout [B,S,H*D]
    gemm(Ob, w[3], Ocat, wb[3], S_, D_, D_, D_, D_, H_ * D_,
         (long long)H_ * S_ * D_, (long long)S_ * D_, 0, (long long)D_ * D_,
         (long long)S_ * H_ * D_, (long long)D_, H_, B_ * H_, D_, 1.0f,
         FLAG_BIAS | FLAG_OBF);
    // final projection cat @ wf + bf
    gemm(Ocat, w[4], attnF, wb[4], B_ * S_, D_, H_ * D_, H_ * D_, H_ * D_, D_,
         0, 0, 0, 0, 0, 0, 1, 1, 0, 1.0f, FLAG_BIAS);
    add_ln<<<B_ * S_, 256, 0, stream>>>(attnF, resid, lng, lnb, outF, outBf, D_);
  };

  // --- self-attention (causal) -> x1 ---
  {
    __bf16* w[5]  = {wT[0], wT[1], wT[2], wT[3], wT[4]};
    const float* wb[5] = {saW[1], saW[3], saW[5], saW[7], saW[9]};
    mha(xbf, xbf, T_, w, wb, 1, x_f, ln1g, ln1b, x1F, x1bf);
  }
  // --- cross-attention (no mask) -> x2 ---
  {
    __bf16* w[5]  = {wT[5], wT[6], wT[7], wT[8], wT[9]};
    const float* wb[5] = {caW[1], caW[3], caW[5], caW[7], caW[9]};
    mha(x1bf, encbf, T_, w, wb, 0, x1F, ln2g, ln2b, x2F, x2bf);
  }
  // --- FFN ---
  gemm(x2bf, fc1T, ff1, fc1b, B_ * S_, 4 * D_, D_, D_, D_, 4 * D_,
       0, 0, 0, 0, 0, 0, 1, 1, 0, 1.0f, FLAG_BIAS | FLAG_RELU | FLAG_OBF);
  gemm(ff1, fc2T, ff2, fc2b, B_ * S_, D_, 4 * D_, 4 * D_, 4 * D_, D_,
       0, 0, 0, 0, 0, 0, 1, 1, 0, 1.0f, FLAG_BIAS);
  add_ln<<<B_ * S_, 256, 0, stream>>>(ff2, x2F, ln3g, ln3b, (float*)d_out, nullptr, D_);
}
